// FaceResVQ_86870008529645
// MI455X (gfx1250) — matrix operand order repaired and checked by reading, not compiled, not run on hardware
//
#include <hip/hip_runtime.h>

// ---------------------------------------------------------------------------
// FaceResVQ on MI455X (gfx1250): 2-stage residual VQ, codebook in LDS (bf16),
// distance GEMM on v_wmma_f32_16x16x32_bf16 (2 column-tiles in flight to fill
// WMMA hazard slots with the other tile's argmin VALU), exact f32 epilogue.
// ---------------------------------------------------------------------------

typedef __attribute__((ext_vector_type(16))) __bf16 v16bf;
typedef __attribute__((ext_vector_type(8)))  float  v8f;

#define NROWS        196608          // B * NF * V = 4 * 16384 * 3
#define DV           64
#define KCB          1024
#define NTILES       (NROWS / 16)    // 12288 sixteen-row tiles
#define NSTAGES      2               // QPV
#define CB_STRIDE_U  36              // padded LDS row stride in dwords (72 bf16) -> conflict-free
#define CBSQ_OFF     (KCB * CB_STRIDE_U)
#define LDS_BYTES    ((KCB * CB_STRIDE_U + KCB) * 4)   // 151552 B (< 320 KB/WGP)
#define QUANT_ELEMS  (NROWS * DV)    // 12582912
#define IDX_ELEMS    (NROWS * NSTAGES)
#define GRID_WGS     512
#define INV_TOTAL    (1.0f / 12582912.0f)

// round-to-nearest-even f32 -> bf16 (bit container only)
static __device__ __forceinline__ unsigned f2bf(float f) {
  unsigned u = __builtin_bit_cast(unsigned, f);
  return (u + 0x7FFFu + ((u >> 16) & 1u)) >> 16;
}
static __device__ __forceinline__ unsigned packbf(float lo, float hi) {
  return f2bf(lo) | (f2bf(hi) << 16);
}

struct U8box { uint4 lo, hi; };   // 32 bytes == v16bf

static __device__ __forceinline__ v16bf ld_b16x16_lds(const unsigned* p) {
  U8box t;
  t.lo = *(const uint4*)(p);
  t.hi = *(const uint4*)(p + 8);
  return __builtin_bit_cast(v16bf, t);
}

__global__ void rvq_init(float* __restrict__ loss) {
  loss[0] = 0.0f;
  loss[1] = 0.0f;
}

__global__ __launch_bounds__(256) void rvq_main(const float* __restrict__ x,
                                                const float* __restrict__ cb,
                                                float* __restrict__ out) {
  extern __shared__ unsigned smem[];           // [KCB][36] packed bf16 codebook
  float* cbsq = (float*)(smem + CBSQ_OFF);     // [KCB] ||c||^2 (f32)

  const int tid = threadIdx.x;

  // ---- Stage codebook into LDS: f32 -> bf16 (padded rows) + squared norms ----
  for (int row = tid; row < KCB; row += 256) {
    const float4* src = (const float4*)(cb + row * DV);
    unsigned* dst = smem + row * CB_STRIDE_U;
    float ss = 0.0f;
#pragma unroll
    for (int j = 0; j < 16; ++j) {
      float4 v = src[j];
      ss += v.x * v.x + v.y * v.y + v.z * v.z + v.w * v.w;
      dst[2 * j]     = packbf(v.x, v.y);
      dst[2 * j + 1] = packbf(v.z, v.w);
    }
    cbsq[row] = ss;
  }
  __syncthreads();

  const int lane   = tid & 31;
  const int wave   = tid >> 5;
  const int gwave  = blockIdx.x * 8 + wave;
  const int nwaves = gridDim.x * 8;

  // ISA 16-bit A layout: low lanes own K {0..7,16..23}(+32), high lanes {8..15,24..31}(+32)
  const int kbase   = (lane >= 16) ? 8 : 0;
  const int khalf   = lane >> 4;        // B-operand K half
  const int colLane = lane & 15;        // B/C column within a 16-wide tile

  int*   idxOut  = ((int*)out) + QUANT_ELEMS;
  float* lossOut = out + QUANT_ELEMS + IDX_ELEMS;

  float lossAcc0 = 0.0f, lossAcc1 = 0.0f;

  for (int tile = gwave; tile < NTILES; tile += nwaves) {
    const int row = tile * 16 + (lane & 15);

    // ---- load residual tile in WMMA-A element order; chunk c -> K run of 4 ----
    float r[32], quant[32];
#pragma unroll
    for (int c = 0; c < 8; ++c) {
      const int h = c >> 2, cc = c & 3;
      const int k0 = h * 32 + kbase + ((cc >> 1) ? 16 : 0) + (cc & 1) * 4;
      const int e0 = h * 16 + cc * 4;
      float4 v = *(const float4*)(x + row * DV + k0);
      r[e0] = v.x; r[e0 + 1] = v.y; r[e0 + 2] = v.z; r[e0 + 3] = v.w;
      quant[e0] = 0.0f; quant[e0 + 1] = 0.0f; quant[e0 + 2] = 0.0f; quant[e0 + 3] = 0.0f;
    }

#pragma unroll
    for (int stage = 0; stage < NSTAGES; ++stage) {
      // ---- A operands: bf16(-2 * residual), folding the -2 of the distance ----
      unsigned aw[16];
#pragma unroll
      for (int j = 0; j < 16; ++j)
        aw[j] = packbf(-2.0f * r[2 * j], -2.0f * r[2 * j + 1]);
      U8box t0, t1;
      t0.lo = make_uint4(aw[0], aw[1], aw[2], aw[3]);
      t0.hi = make_uint4(aw[4], aw[5], aw[6], aw[7]);
      t1.lo = make_uint4(aw[8], aw[9], aw[10], aw[11]);
      t1.hi = make_uint4(aw[12], aw[13], aw[14], aw[15]);
      v16bf a0 = __builtin_bit_cast(v16bf, t0);
      v16bf a1 = __builtin_bit_cast(v16bf, t1);

      float mv[8]; int mi[8];
#pragma unroll
      for (int i = 0; i < 8; ++i) { mv[i] = 3.4e38f; mi[i] = 0; }

      // ---- score GEMM: s = ||c||^2 - 2 r.c ; two column tiles in flight so the
      //      argmin VALU of tile A co-executes under tile B's WMMAs -------------
      for (int t = 0; t < 64; t += 2) {
        const int colA = t * 16 + colLane;
        const int colB = colA + 16;
        const unsigned* bpA = smem + colA * CB_STRIDE_U + khalf * 4;
        const unsigned* bpB = bpA + 16 * CB_STRIDE_U;
        v16bf b0A = ld_b16x16_lds(bpA);
        v16bf b1A = ld_b16x16_lds(bpA + 16);
        v16bf b0B = ld_b16x16_lds(bpB);
        v16bf b1B = ld_b16x16_lds(bpB + 16);
        const float cbsA = cbsq[colA];
        const float cbsB = cbsq[colB];
        v8f accA, accB;
#pragma unroll
        for (int i = 0; i < 8; ++i) { accA[i] = cbsA; accB[i] = cbsB; }

        // interleave the two independent accumulator chains
        accA = __builtin_amdgcn_wmma_f32_16x16x32_bf16(false, a0, false, b0A,
                                                       (short)0, accA, false, false);
        accB = __builtin_amdgcn_wmma_f32_16x16x32_bf16(false, a0, false, b0B,
                                                       (short)0, accB, false, false);
        accA = __builtin_amdgcn_wmma_f32_16x16x32_bf16(false, a1, false, b1A,
                                                       (short)0, accA, false, false);
        accB = __builtin_amdgcn_wmma_f32_16x16x32_bf16(false, a1, false, b1B,
                                                       (short)0, accB, false, false);

#pragma unroll
        for (int i = 0; i < 8; ++i) {
          const float sA = accA[i];
          const bool bA = sA < mv[i];
          mi[i] = bA ? colA : mi[i];
          mv[i] = bA ? sA : mv[i];
        }
#pragma unroll
        for (int i = 0; i < 8; ++i) {
          const float sB = accB[i];
          const bool bB = sB < mv[i];
          mi[i] = bB ? colB : mi[i];
          mv[i] = bB ? sB : mv[i];
        }
      }

      // ---- branchless argmin across the 16 lanes sharing each row ----
#pragma unroll
      for (int m = 1; m < 16; m <<= 1) {
#pragma unroll
        for (int i = 0; i < 8; ++i) {
          const float ov = __shfl_xor(mv[i], m, 32);
          const int   oi = __shfl_xor(mi[i], m, 32);
          const bool take = (ov < mv[i]) | ((ov == mv[i]) & (oi < mi[i]));
          mv[i] = take ? ov : mv[i];
          mi[i] = take ? oi : mi[i];
        }
      }

      // slot i holds row i (low lanes) / row 8+i (high lanes); fetch my row's idx
      const int s8 = lane & 7;
      int own = mi[0];
#pragma unroll
      for (int i = 1; i < 8; ++i) own = (s8 == i) ? mi[i] : own;
      const int  other    = __shfl_xor(own, 16, 32);
      const bool haveHigh = lane >= 16;
      const bool wantHigh = (lane & 15) >= 8;
      const int  rowIdx   = (haveHigh == wantHigh) ? own : other;

      if (lane < 16) idxOut[row * NSTAGES + stage] = rowIdx;

      // ---- exact f32: gather code row, commit loss, residual & quant update ----
      float lsum = 0.0f;
#pragma unroll
      for (int c = 0; c < 8; ++c) {
        const int h = c >> 2, cc = c & 3;
        const int k0 = h * 32 + kbase + ((cc >> 1) ? 16 : 0) + (cc & 1) * 4;
        const int e0 = h * 16 + cc * 4;
        float4 q = *(const float4*)(cb + rowIdx * DV + k0);   // 256 KB, L2-hot
        const float d0 = q.x - r[e0],     d1 = q.y - r[e0 + 1];
        const float d2 = q.z - r[e0 + 2], d3 = q.w - r[e0 + 3];
        lsum += d0 * d0 + d1 * d1 + d2 * d2 + d3 * d3;
        quant[e0]     += q.x; quant[e0 + 1] += q.y;
        quant[e0 + 2] += q.z; quant[e0 + 3] += q.w;
        r[e0]     = -d0; r[e0 + 1] = -d1;   // r <- r - q
        r[e0 + 2] = -d2; r[e0 + 3] = -d3;
      }
      if (stage == 0) lossAcc0 += lsum; else lossAcc1 += lsum;
    }

    // ---- store quantized = q1 + q2 (flat layout == reference reshape) ----
#pragma unroll
    for (int c = 0; c < 8; ++c) {
      const int h = c >> 2, cc = c & 3;
      const int k0 = h * 32 + kbase + ((cc >> 1) ? 16 : 0) + (cc & 1) * 4;
      const int e0 = h * 16 + cc * 4;
      float4 v;
      v.x = quant[e0]; v.y = quant[e0 + 1]; v.z = quant[e0 + 2]; v.w = quant[e0 + 3];
      *(float4*)(out + row * DV + k0) = v;
    }
  }

  // ---- commit-loss flush: wave reduce then one atomic per stage per wave ----
#pragma unroll
  for (int m = 1; m < 32; m <<= 1) {
    lossAcc0 += __shfl_xor(lossAcc0, m, 32);
    lossAcc1 += __shfl_xor(lossAcc1, m, 32);
  }
  if (lane == 0) {
    unsafeAtomicAdd(&lossOut[0], lossAcc0 * INV_TOTAL);
    unsafeAtomicAdd(&lossOut[1], lossAcc1 * INV_TOTAL);
  }
}

extern "C" void kernel_launch(void* const* d_in, const int* in_sizes, int n_in,
                              void* d_out, int out_size, void* d_ws, size_t ws_size,
                              hipStream_t stream) {
  (void)in_sizes; (void)n_in; (void)d_ws; (void)ws_size; (void)out_size;
  const float* x   = (const float*)d_in[0];   // (4,16384,192) f32
  const float* cb  = (const float*)d_in[1];   // (1024,64) f32
  float*       out = (float*)d_out;           // [quant | idx | loss]

  rvq_init<<<1, 1, 0, stream>>>(out + QUANT_ELEMS + IDX_ELEMS);
  rvq_main<<<GRID_WGS, 256, LDS_BYTES, stream>>>(x, cb, out);
}
